// VQVAE_52630529245201
// MI455X (gfx1250) — matrix operand order repaired and checked by reading, not compiled, not run on hardware
//
#include <hip/hip_runtime.h>
#include <hip/hip_bf16.h>

typedef __attribute__((ext_vector_type(16))) _Float16 v16h;
typedef __attribute__((ext_vector_type(8)))  float    v8f;

// K index this file uses for 16-bit A/B fragments: contiguous, k = 16*half + j
__device__ __forceinline__ int a_kmap(int half, int j) {
  return half * 16 + j;
}

// ---------------------------------------------------------------------------
// Pack conv weights [Co][Ktot] f32 into WMMA A-fragment order, zero-padded in
// both K (to nkc*32) and Co (to ntiles*16):
//   Wp[(((tile*nkc)+kc)*32 + lane)*16 + j],  kd = kc*32 + a_kmap(lane>>4, j)
// ---------------------------------------------------------------------------
__global__ void pack_weights(const float* __restrict__ W, _Float16* __restrict__ Wp,
                             int Co, int Ktot, int nkc, int ntiles) {
  int total = ntiles * nkc * 512;
  int idx = blockIdx.x * 256 + threadIdx.x;
  if (idx >= total) return;
  int j    = idx & 15;
  int lane = (idx >> 4) & 31;
  int kc   = (idx >> 9) % nkc;
  int tile = idx / (nkc * 512);
  int kd   = kc * 32 + a_kmap(lane >> 4, j);
  int co   = tile * 16 + (lane & 15);
  float v = 0.f;
  if (co < Co && kd < Ktot) v = W[(size_t)co * Ktot + kd];
  Wp[idx] = (_Float16)v;
}

// ---------------------------------------------------------------------------
// Input x -> f16, halo-padded layout [B][Cipad][T+2] (zero halo cols + zero
// extra channels so downstream im2col needs no guards).
// ---------------------------------------------------------------------------
__global__ void cvt_x_padded(const float* __restrict__ x, _Float16* __restrict__ xp,
                             int Cin, int Cipad, int T, int B) {
  const int Tpad = T + 2;
  size_t total = (size_t)B * Cipad * Tpad;
  for (size_t i = (size_t)blockIdx.x * blockDim.x + threadIdx.x; i < total;
       i += (size_t)gridDim.x * blockDim.x) {
    int    tp  = (int)(i % Tpad);
    size_t row = i / Tpad;
    int    ci  = (int)(row % Cipad);
    int    n   = (int)(row / Cipad);
    _Float16 v = (_Float16)0.0f;
    if (ci < Cin && tp >= 1 && tp <= T)
      v = (_Float16)x[((size_t)n * Cin + ci) * T + (tp - 1)];
    xp[i] = v;
  }
}

// ---------------------------------------------------------------------------
// im2col: halo-padded [B][Cipad][T+2] f16 -> K-contiguous [B][T][Kpad] f16.
// Guard-free: Cipad = ceil(Kpad/KW), halo covers t edges.
// ---------------------------------------------------------------------------
template <int KW>
__global__ void im2col_f16(const _Float16* __restrict__ Xp,
                           _Float16* __restrict__ Xim,
                           int Cipad, int T, int B, int Kpad, int pad) {
  const int Tpad = T + 2;
  size_t total = (size_t)B * T * Kpad;
  for (size_t i = (size_t)blockIdx.x * blockDim.x + threadIdx.x; i < total;
       i += (size_t)gridDim.x * blockDim.x) {
    int    kd = (int)(i % Kpad);
    size_t rt = i / Kpad;
    int    t  = (int)(rt % T);
    int    n  = (int)(rt / T);
    int    ci = kd / KW;                       // compile-time divisor
    int    kw = kd - ci * KW;
    Xim[i] = Xp[((size_t)n * Cipad + ci) * Tpad + (t + kw - pad + 1)];
  }
}

// ---------------------------------------------------------------------------
// Implicit-GEMM conv1d via WMMA, all-vector-load K-loop, 2-way Co blocking.
//   Wp : packed A fragments (ntiles*nkc chunks)
//   Xim: [B][T][Kpad] f16 (K-contiguous im2col)
//   Y  : [B][Co][T] f32 dense (bias added)
// grid = (T/64, ntiles/2, B), block = 128 (4 waves, 32x16 tile each).
// ---------------------------------------------------------------------------
__global__ void conv1d_wmma2(const _Float16* __restrict__ Wp,
                             const _Float16* __restrict__ Xim,
                             const float* __restrict__ bias,
                             float* __restrict__ Y,
                             int Co, int T, int nkc) {
  const int lane = threadIdx.x & 31;
  const int wave = threadIdx.x >> 5;
  const int half = lane >> 4;
  const int l16  = lane & 15;
  const int n    = blockIdx.z;
  const int tile0 = blockIdx.y * 2;
  const int t0   = blockIdx.x * 64 + wave * 16;
  const int Kpad = nkc * 32;

  const _Float16* wp0 = Wp + ((size_t)tile0 * nkc * 32 + lane) * 16;
  const _Float16* wp1 = wp0 + (size_t)nkc * 512;
  const _Float16* xi  = Xim + ((size_t)n * T + t0 + l16) * Kpad + half * 16;

  v8f acc0 = {}, acc1 = {};
  for (int kc = 0; kc < nkc; ++kc) {
    v16h b  = *(const v16h*)xi;  xi  += 32;
    v16h a0 = *(const v16h*)wp0; wp0 += 512;
    v16h a1 = *(const v16h*)wp1; wp1 += 512;
    acc0 = __builtin_amdgcn_wmma_f32_16x16x32_f16(false, a0, false, b,
                                                  (short)0, acc0, false, false);
    acc1 = __builtin_amdgcn_wmma_f32_16x16x32_f16(false, a1, false, b,
                                                  (short)0, acc1, false, false);
  }

#pragma unroll
  for (int r = 0; r < 8; ++r) {
    int co = tile0 * 16 + r + half * 8;
    if (co < Co)
      Y[((size_t)n * Co + co) * T + (t0 + l16)] = acc0[r] + bias[co];
    int co1 = co + 16;
    if (co1 < Co)
      Y[((size_t)n * Co + co1) * T + (t0 + l16)] = acc1[r] + bias[co1];
  }
}

// ---------------------------------------------------------------------------
// BatchNorm training-mode stats per channel: mean + rsqrt(var+eps).
// ---------------------------------------------------------------------------
__global__ void bn_stats(const float* __restrict__ x, float* __restrict__ stats,
                         int C, int T, int B) {
  __shared__ float s1[256], s2[256];
  const int c = blockIdx.x;
  const int NT = B * T;
  float a = 0.f, q = 0.f;
  for (int i = threadIdx.x; i < NT; i += 256) {
    int n = i / T, t = i - n * T;
    float v = x[((size_t)n * C + c) * T + t];
    a += v; q += v * v;
  }
  s1[threadIdx.x] = a; s2[threadIdx.x] = q;
  __syncthreads();
  for (int s = 128; s > 0; s >>= 1) {
    if (threadIdx.x < s) {
      s1[threadIdx.x] += s1[threadIdx.x + s];
      s2[threadIdx.x] += s2[threadIdx.x + s];
    }
    __syncthreads();
  }
  if (threadIdx.x == 0) {
    float mean = s1[0] / NT;
    float var  = s2[0] / NT - mean * mean;
    stats[c]     = mean;
    stats[C + c] = rsqrtf(var + 1e-5f);
  }
}

// ---------------------------------------------------------------------------
// BN + ReLU -> f16 halo-padded layout [B][C][T+2] (halo columns written 0).
// ---------------------------------------------------------------------------
__global__ void bn_apply_relu_pad(const float* __restrict__ x,
                                  const float* __restrict__ stats,
                                  const float* __restrict__ g,
                                  const float* __restrict__ b,
                                  _Float16* __restrict__ yp,
                                  int C, int T, int B) {
  const int Tpad = T + 2;
  size_t total = (size_t)B * C * Tpad;
  for (size_t i = (size_t)blockIdx.x * blockDim.x + threadIdx.x; i < total;
       i += (size_t)gridDim.x * blockDim.x) {
    int    tp  = (int)(i % Tpad);
    size_t row = i / Tpad;
    int    c   = (int)(row % C);
    _Float16 o = (_Float16)0.0f;
    if (tp >= 1 && tp <= T) {
      float v = (x[row * T + (tp - 1)] - stats[c]) * stats[C + c] * g[c] + b[c];
      o = (_Float16)(v > 0.f ? v : 0.f);
    }
    yp[i] = o;
  }
}

// ---------------------------------------------------------------------------
// Codebook prep: f16 copy + squared norms (512 x 64).
// ---------------------------------------------------------------------------
__global__ void cb_prep(const float* __restrict__ cb,
                        _Float16* __restrict__ cb16,
                        float* __restrict__ cnorm) {
  int i = blockIdx.x * blockDim.x + threadIdx.x;
  if (i < 512) {
    float s = 0.f;
    for (int d = 0; d < 64; ++d) {
      float v = cb[i * 64 + d];
      s += v * v;
      cb16[i * 64 + d] = (_Float16)v;
    }
    cnorm[i] = s;
  }
}

// ---------------------------------------------------------------------------
// Pack z [B][64][T] f32 into WMMA A-fragment order (f16).
// ---------------------------------------------------------------------------
__global__ void pack_z(const float* __restrict__ z, _Float16* __restrict__ zp,
                       int T, int B) {
  int ntile = T / 16;
  int total = B * ntile * 1024;
  int idx = blockIdx.x * 256 + threadIdx.x;
  if (idx >= total) return;
  int j0   = idx & 31;
  int lane = (idx >> 5) & 31;
  int rest = idx >> 10;
  int tile = rest % ntile;
  int n    = rest / ntile;
  int j    = j0 & 15;
  int fr   = j0 >> 4;
  int kd   = a_kmap(lane >> 4, j) + fr * 32;
  int t    = tile * 16 + (lane & 15);
  zp[idx] = (_Float16)z[((size_t)n * 64 + kd) * T + t];
}

// ---------------------------------------------------------------------------
// Zero halo columns of a padded f16 buffer (rows x (T+2)).
// ---------------------------------------------------------------------------
__global__ void zero_halo(_Float16* __restrict__ buf, int rows, int Tpad) {
  int idx = blockIdx.x * 256 + threadIdx.x;
  if (idx < rows * 2) {
    int row = idx >> 1, side = idx & 1;
    buf[(size_t)row * Tpad + (size_t)side * (Tpad - 1)] = (_Float16)0.0f;
  }
}

// ---------------------------------------------------------------------------
// VQ argmin via WMMA: score = ||c||^2 - 2 z.c; all fragment loads are aligned
// v16h vector loads. grid = (T/64, B), block = 128.
// ---------------------------------------------------------------------------
__global__ void vq_argmin_wmma(const _Float16* __restrict__ zp,
                               const _Float16* __restrict__ cb16,
                               const float* __restrict__ cnorm,
                               const float* __restrict__ cbf32,
                               float* __restrict__ qf32,      // [B][64][T] dense
                               _Float16* __restrict__ qf16p,  // [B][64][T+2]
                               float* __restrict__ idx_out,   // [B][T]
                               int T) {
  const int lane = threadIdx.x & 31;
  const int wave = threadIdx.x >> 5;
  const int half = lane >> 4;
  const int l16  = lane & 15;
  const int n    = blockIdx.y;
  const int tile = blockIdx.x * 4 + wave;
  const int Tpad = T + 2;

  const _Float16* za = zp + (((size_t)n * (T / 16) + tile) * 32 + lane) * 32;
  v16h a0 = *(const v16h*)za;
  v16h a1 = *(const v16h*)(za + 16);

  float best[8];
  int   bidx[8];
#pragma unroll
  for (int r = 0; r < 8; ++r) { best[r] = 3.0e38f; bidx[r] = 0; }

  for (int c0 = 0; c0 < 512; c0 += 16) {
    const int code = c0 + l16;
    const _Float16* cbp = cb16 + (size_t)code * 64 + half * 16;
    v16h b0 = *(const v16h*)cbp;
    v16h b1 = *(const v16h*)(cbp + 32);
    v8f acc = {};
    acc = __builtin_amdgcn_wmma_f32_16x16x32_f16(false, a0, false, b0,
                                                 (short)0, acc, false, false);
    acc = __builtin_amdgcn_wmma_f32_16x16x32_f16(false, a1, false, b1,
                                                 (short)0, acc, false, false);
    float cn = cnorm[code];
#pragma unroll
    for (int r = 0; r < 8; ++r) {
      float score = cn - 2.0f * acc[r];
      if (score < best[r]) { best[r] = score; bidx[r] = code; }
    }
  }

  __shared__ float ss[4][16][16];
  __shared__ int   si[4][16][16];
#pragma unroll
  for (int r = 0; r < 8; ++r) {
    int m = r + half * 8;
    ss[wave][m][l16] = best[r];
    si[wave][m][l16] = bidx[r];
  }
  __syncthreads();

  const int tt = threadIdx.x;
  if (tt < 64) {
    int w = tt >> 4, m = tt & 15;
    float bs = ss[w][m][0];
    int   bc = si[w][m][0];
    for (int j = 1; j < 16; ++j) {
      float s = ss[w][m][j];
      int   c = si[w][m][j];
      if (s < bs || (s == bs && c < bc)) { bs = s; bc = c; }
    }
    int t = blockIdx.x * 64 + w * 16 + m;
    idx_out[(size_t)n * T + t] = (float)bc;
    for (int d = 0; d < 64; ++d) {
      float v = cbf32[(size_t)bc * 64 + d];
      qf32[((size_t)n * 64 + d) * T + t] = v;
      qf16p[((size_t)n * 64 + d) * Tpad + (t + 1)] = (_Float16)v;
    }
  }
}

// ---------------------------------------------------------------------------
// vq_loss = 1.25 * mean((q-z)^2): deterministic two-stage reduction.
// ---------------------------------------------------------------------------
__global__ void sqdiff_partial(const float* __restrict__ q,
                               const float* __restrict__ z,
                               float* __restrict__ partials, size_t n) {
  __shared__ float s[256];
  float a = 0.f;
  for (size_t i = (size_t)blockIdx.x * 256 + threadIdx.x; i < n;
       i += (size_t)gridDim.x * 256) {
    float d = q[i] - z[i];
    a += d * d;
  }
  s[threadIdx.x] = a;
  __syncthreads();
  for (int st = 128; st > 0; st >>= 1) {
    if (threadIdx.x < st) s[threadIdx.x] += s[threadIdx.x + st];
    __syncthreads();
  }
  if (threadIdx.x == 0) partials[blockIdx.x] = s[0];
}

__global__ void loss_final(const float* __restrict__ partials, int nb,
                           float* __restrict__ out, float scale) {
  __shared__ float s[256];
  float a = 0.f;
  for (int i = threadIdx.x; i < nb; i += 256) a += partials[i];
  s[threadIdx.x] = a;
  __syncthreads();
  for (int st = 128; st > 0; st >>= 1) {
    if (threadIdx.x < st) s[threadIdx.x] += s[threadIdx.x + st];
    __syncthreads();
  }
  if (threadIdx.x == 0) out[0] = s[0] * scale;
}

// ---------------------------------------------------------------------------
// Host-side launch
// ---------------------------------------------------------------------------
extern "C" void kernel_launch(void* const* d_in, const int* in_sizes, int n_in,
                              void* d_out, int out_size, void* d_ws, size_t ws_size,
                              hipStream_t stream) {
  (void)in_sizes; (void)n_in; (void)out_size; (void)ws_size;

  constexpr int B = 16, Cin = 80, T = 4096, H1 = 128, H2 = 256, D = 64, K = 512;
  constexpr int Tpad = T + 2;
  constexpr size_t NX   = (size_t)B * Cin * T;
  constexpr size_t NH1  = (size_t)B * H1 * T;
  constexpr size_t NH2  = (size_t)B * H2 * T;
  constexpr size_t NBDT = (size_t)B * D * T;
  // per-conv geometry: nkc = ceil(Ktot/32), Cipad = ceil(nkc*32/Kw),
  // ntiles = Co padded to multiple of 32 (2-way blocking) / 16
  constexpr int NKC1 = 8,  CIP1 = 86,  NT1 = 8;   // enc1: K=240, Co=128
  constexpr int NKC2 = 12, CIP2 = 128, NT2 = 16;  // enc2: K=384, Co=256
  constexpr int NKC3 = 8,  CIP3 = 256, NT3 = 4;   // enc3: K=256, Co=64
  constexpr int NKD1 = 6,  CID1 = 64,  ND1 = 16;  // dec1: K=192, Co=256
  constexpr int NKD2 = 24, CID2 = 256, ND2 = 8;   // dec2: K=768, Co=128
  constexpr int NKD3 = 12, CID3 = 128, ND3 = 6;   // dec3: K=384, Co=80->96

  const float* x      = (const float*)d_in[0];
  const float* enc_w1 = (const float*)d_in[1];
  const float* enc_b1 = (const float*)d_in[2];
  const float* bn1_g  = (const float*)d_in[3];
  const float* bn1_b  = (const float*)d_in[4];
  const float* enc_w2 = (const float*)d_in[5];
  const float* enc_b2 = (const float*)d_in[6];
  const float* bn2_g  = (const float*)d_in[7];
  const float* bn2_b  = (const float*)d_in[8];
  const float* enc_w3 = (const float*)d_in[9];
  const float* enc_b3 = (const float*)d_in[10];
  const float* cbf32  = (const float*)d_in[11];
  const float* dec_w1 = (const float*)d_in[12];
  const float* dec_b1 = (const float*)d_in[13];
  const float* dbn1_g = (const float*)d_in[14];
  const float* dbn1_b = (const float*)d_in[15];
  const float* dec_w2 = (const float*)d_in[16];
  const float* dec_b2 = (const float*)d_in[17];
  const float* dbn2_g = (const float*)d_in[18];
  const float* dbn2_b = (const float*)d_in[19];
  const float* dec_w3 = (const float*)d_in[20];
  const float* dec_b3 = (const float*)d_in[21];

  float* out = (float*)d_out;
  float* out_recon = out;            // [B][Cin][T]
  float* out_loss  = out + NX;       // [1]
  float* out_idx   = out + NX + 1;   // [B][T]

  char* ws = (char*)d_ws;
  size_t off = 0;
  auto take = [&](size_t bytes) -> char* {
    char* p = ws + off;
    off += (bytes + 255) & ~(size_t)255;
    return p;
  };
  _Float16* x16p  = (_Float16*)take((size_t)B * CIP1 * Tpad * 2);
  _Float16* w1p   = (_Float16*)take((size_t)NT1 * NKC1 * 512 * 2);
  _Float16* w2p   = (_Float16*)take((size_t)NT2 * NKC2 * 512 * 2);
  _Float16* w3p   = (_Float16*)take((size_t)NT3 * NKC3 * 512 * 2);
  _Float16* dw1p  = (_Float16*)take((size_t)ND1 * NKD1 * 512 * 2);
  _Float16* dw2p  = (_Float16*)take((size_t)ND2 * NKD2 * 512 * 2);
  _Float16* dw3p  = (_Float16*)take((size_t)ND3 * NKD3 * 512 * 2);
  _Float16* cb16  = (_Float16*)take((size_t)K * D * 2);
  float*    cnorm = (float*)take(K * 4);
  float*    stats = (float*)take(2 * 256 * 4);
  float*    parts = (float*)take(256 * 4);
  float*    zf32  = (float*)take(NBDT * 4);
  _Float16* zpk   = (_Float16*)take(NBDT * 2);
  float*    qf32  = (float*)take(NBDT * 4);
  _Float16* qf16p = (_Float16*)take((size_t)B * D * Tpad * 2);
  float*    rawA  = (float*)take(NH1 * 4);                      // h1raw / d2raw
  float*    rawB  = (float*)take(NH2 * 4);                      // h2raw / d1raw
  _Float16* f16A  = (_Float16*)take((size_t)B * H1 * Tpad * 2); // h1f16 / d2f16
  _Float16* f16B  = (_Float16*)take((size_t)B * H2 * Tpad * 2); // h2f16 / d1f16
  _Float16* imb   = (_Float16*)take((size_t)B * T * (NKD2 * 32) * 2); // im2col (max)

  const int EB = 2048, ET = 256;
  auto packw = [&](const float* src, _Float16* dst, int Co, int Ktot, int nkc,
                   int ntiles) {
    int total = ntiles * nkc * 512;
    pack_weights<<<(total + 255) / 256, 256, 0, stream>>>(src, dst, Co, Ktot,
                                                          nkc, ntiles);
  };

  // ---- prep ----
  cvt_x_padded<<<EB, ET, 0, stream>>>(x, x16p, Cin, CIP1, T, B);
  packw(enc_w1, w1p,  H1,  Cin * 3, NKC1, NT1);
  packw(enc_w2, w2p,  H2,  H1 * 3,  NKC2, NT2);
  packw(enc_w3, w3p,  D,   H2,      NKC3, NT3);
  packw(dec_w1, dw1p, H2,  D * 3,   NKD1, ND1);
  packw(dec_w2, dw2p, H1,  H2 * 3,  NKD2, ND2);
  packw(dec_w3, dw3p, Cin, H1 * 3,  NKD3, ND3);
  cb_prep<<<2, 256, 0, stream>>>(cbf32, cb16, cnorm);

  // ---- encoder ----
  im2col_f16<3><<<EB, ET, 0, stream>>>(x16p, imb, CIP1, T, B, NKC1 * 32, 1);
  conv1d_wmma2<<<dim3(T / 64, NT1 / 2, B), 128, 0, stream>>>(
      w1p, imb, enc_b1, rawA, H1, T, NKC1);
  bn_stats<<<H1, 256, 0, stream>>>(rawA, stats, H1, T, B);
  bn_apply_relu_pad<<<EB, ET, 0, stream>>>(rawA, stats, bn1_g, bn1_b, f16A, H1, T, B);

  im2col_f16<3><<<EB, ET, 0, stream>>>(f16A, imb, CIP2, T, B, NKC2 * 32, 1);
  conv1d_wmma2<<<dim3(T / 64, NT2 / 2, B), 128, 0, stream>>>(
      w2p, imb, enc_b2, rawB, H2, T, NKC2);
  bn_stats<<<H2, 256, 0, stream>>>(rawB, stats, H2, T, B);
  bn_apply_relu_pad<<<EB, ET, 0, stream>>>(rawB, stats, bn2_g, bn2_b, f16B, H2, T, B);

  im2col_f16<1><<<EB, ET, 0, stream>>>(f16B, imb, CIP3, T, B, NKC3 * 32, 0);
  conv1d_wmma2<<<dim3(T / 64, NT3 / 2, B), 128, 0, stream>>>(
      w3p, imb, enc_b3, zf32, D, T, NKC3);

  // ---- vector quantizer ----
  {
    int total = B * (T / 16) * 1024;
    pack_z<<<(total + 255) / 256, 256, 0, stream>>>(zf32, zpk, T, B);
  }
  zero_halo<<<(B * D * 2 + 255) / 256, 256, 0, stream>>>(qf16p, B * D, Tpad);
  vq_argmin_wmma<<<dim3(T / 64, B), 128, 0, stream>>>(
      zpk, cb16, cnorm, cbf32, qf32, qf16p, out_idx, T);
  sqdiff_partial<<<256, 256, 0, stream>>>(qf32, zf32, parts, NBDT);
  loss_final<<<1, 256, 0, stream>>>(parts, 256, out_loss, 1.25f / (float)NBDT);

  // ---- decoder ----
  im2col_f16<3><<<EB, ET, 0, stream>>>(qf16p, imb, CID1, T, B, NKD1 * 32, 1);
  conv1d_wmma2<<<dim3(T / 64, ND1 / 2, B), 128, 0, stream>>>(
      dw1p, imb, dec_b1, rawB, H2, T, NKD1);
  bn_stats<<<H2, 256, 0, stream>>>(rawB, stats, H2, T, B);
  bn_apply_relu_pad<<<EB, ET, 0, stream>>>(rawB, stats, dbn1_g, dbn1_b, f16B, H2, T, B);

  im2col_f16<3><<<EB, ET, 0, stream>>>(f16B, imb, CID2, T, B, NKD2 * 32, 1);
  conv1d_wmma2<<<dim3(T / 64, ND2 / 2, B), 128, 0, stream>>>(
      dw2p, imb, dec_b2, rawA, H1, T, NKD2);
  bn_stats<<<H1, 256, 0, stream>>>(rawA, stats, H1, T, B);
  bn_apply_relu_pad<<<EB, ET, 0, stream>>>(rawA, stats, dbn2_g, dbn2_b, f16A, H1, T, B);

  im2col_f16<3><<<EB, ET, 0, stream>>>(f16A, imb, CID3, T, B, NKD3 * 32, 1);
  conv1d_wmma2<<<dim3(T / 64, ND3 / 2, B), 128, 0, stream>>>(
      dw3p, imb, dec_b3, out_recon, Cin, T, NKD3);
}
